// DotProductAttention_32409823216295
// MI455X (gfx1250) — compile-verified, matrix operand order checked by reading, not run on hardware
//
#include <hip/hip_runtime.h>
#include <math.h>

typedef __attribute__((ext_vector_type(16))) __bf16 v16bf;
typedef __attribute__((ext_vector_type(8)))  float  v8f;

#define NB   16      // batch
#define SEQ  2048    // seq len (Q and K)
#define DM   1024    // d_model
#define PD   512     // proj dim

// ---------- bf16 helpers ----------
__device__ __forceinline__ unsigned short f32_bf16_rne(float f) {
  union { float f; unsigned u; } x; x.f = f;
  unsigned u = x.u;
  unsigned r = u + 0x7FFFu + ((u >> 16) & 1u);
  return (unsigned short)(r >> 16);
}
__device__ __forceinline__ __bf16 to_bf16(float f) {
  unsigned short h = f32_bf16_rne(f);
  return __builtin_bit_cast(__bf16, h);
}
__device__ __forceinline__ unsigned pack2(float a, float b) {
  return (unsigned)f32_bf16_rne(a) | ((unsigned)f32_bf16_rne(b) << 16);
}

// ---------- async copy global -> LDS (CDNA5 ASYNCcnt path, with fallback) ----
#if defined(__gfx1250__) && __has_builtin(__builtin_amdgcn_global_load_async_to_lds_b128)
#define HAVE_ASYNC_LDS 1
#else
#define HAVE_ASYNC_LDS 0
#endif

typedef __attribute__((__vector_size__(4 * sizeof(int)))) int i32x4;
typedef __attribute__((address_space(1))) i32x4* as1_v4i;
typedef __attribute__((address_space(3))) i32x4* as3_v4i;

__device__ __forceinline__ void async_cp16(void* lds, const void* g) {
#if HAVE_ASYNC_LDS
  __builtin_amdgcn_global_load_async_to_lds_b128(
      (as1_v4i)(unsigned long long)g, (as3_v4i)(unsigned long long)lds, 0, 0);
#else
  *(uint4*)lds = *(const uint4*)g;
#endif
}
__device__ __forceinline__ void async_wait() {
#if HAVE_ASYNC_LDS
#if __has_builtin(__builtin_amdgcn_s_wait_asynccnt)
  __builtin_amdgcn_s_wait_asynccnt(0);
#else
  asm volatile("s_wait_asynccnt 0" ::: "memory");
#endif
#endif
}

// ---------- WMMA wrapper ----------
__device__ __forceinline__ v8f wmma_bf16(v16bf a, v16bf b, v8f c) {
  return __builtin_amdgcn_wmma_f32_16x16x32_bf16(false, a, false, b, (short)0, c,
                                                 false, false);
}

// Load a 16x32 bf16 fragment (A-layout; same pattern for B with rows indexed
// by N). ISA: lane m=lane&15 holds row m; element i -> K = kk + (i&7) +
// 16*(i>>3) + 8*(lane>>4)  => two contiguous 16B chunks per lane.
__device__ __forceinline__ v16bf load_frag(const __bf16* p0, int rowStride,
                                           int rowBase, int kk) {
  const int lane = threadIdx.x & 31;
  const int hlf  = lane >> 4;
  const __bf16* p = p0 + (long)(rowBase + (lane & 15)) * rowStride + kk + hlf * 8;
  union { v16bf v; uint4 q[2]; } u;
  u.q[0] = *(const uint4*)(p);
  u.q[1] = *(const uint4*)(p + 16);
  return u.v;
}

// ---------- weight convert f32 -> bf16 ----------
__global__ void cvt_f32_bf16(const float* __restrict__ x, __bf16* __restrict__ y,
                             int n4) {
  int i = blockIdx.x * 256 + threadIdx.x;
  if (i < n4) {
    float4 f = ((const float4*)x)[i];
    uint2 p; p.x = pack2(f.x, f.y); p.y = pack2(f.z, f.w);
    ((uint2*)y)[i] = p;
  }
}

// ---------- GEMM: Y[R,N] = X[R,K] * W[N,K]^T, double-buffered LDS ----------
template <typename TIn, bool OUT_F32>
__global__ __launch_bounds__(256) void gemm_xwT_kernel(
    const TIn* __restrict__ X, const __bf16* __restrict__ W, void* __restrict__ Yv,
    int R, int Kdim, int N) {
  constexpr int BM = 128, BN_ = 64, BK = 64, PAD = 8, STR = BK + PAD;
  __shared__ __align__(16) __bf16 As[2][BM * STR];   // 2 x 18.4 KB
  __shared__ __align__(16) __bf16 Bs[2][BN_ * STR];  // 2 x 9.2 KB

  const int tid  = threadIdx.x;
  const int wave = tid >> 5, lane = tid & 31;
  const int hlf = lane >> 4, l15 = lane & 15;
  const int waveM = wave >> 1, waveN = wave & 1;      // 4x2 wave grid
  const int mBlock = blockIdx.y * BM;
  const int nBlock = blockIdx.x * BN_;

  auto stage = [&](int buf, int k0) {
    if constexpr (sizeof(TIn) == 4) {  // f32 input: load+convert+ds_store
      #pragma unroll
      for (int c = tid; c < (BM * BK) / 4; c += 256) {
        int row = c >> 4, col4 = (c & 15) * 4;
        float4 f = *(const float4*)((const float*)X + (long)(mBlock + row) * Kdim + k0 + col4);
        uint2 p; p.x = pack2(f.x, f.y); p.y = pack2(f.z, f.w);
        *(uint2*)&As[buf][row * STR + col4] = p;
      }
    } else {  // bf16 input: async 16B copies straight to LDS
      #pragma unroll
      for (int c = tid; c < (BM * BK) / 8; c += 256) {
        int row = c >> 3, col8 = (c & 7) * 8;
        async_cp16(&As[buf][row * STR + col8],
                   (const __bf16*)X + (long)(mBlock + row) * Kdim + k0 + col8);
      }
    }
    #pragma unroll
    for (int c = tid; c < (BN_ * BK) / 8; c += 256) {  // weights: async
      int row = c >> 3, col8 = (c & 7) * 8;
      async_cp16(&Bs[buf][row * STR + col8],
                 W + (long)(nBlock + row) * Kdim + k0 + col8);
    }
  };

  v8f c00 = {}, c01 = {}, c10 = {}, c11 = {};
  const int nk = Kdim / BK;
  stage(0, 0);
  for (int i = 0; i < nk; ++i) {
    async_wait();
    __syncthreads();
    if (i + 1 < nk) stage((i + 1) & 1, (i + 1) * BK);  // overlap next tile
    const __bf16* Ab = &As[i & 1][0];
    const __bf16* Bb = &Bs[i & 1][0];
    #pragma unroll
    for (int kk = 0; kk < BK; kk += 32) {
      v16bf a0 = load_frag(Ab, STR, waveM * 32,      kk);
      v16bf a1 = load_frag(Ab, STR, waveM * 32 + 16, kk);
      v16bf b0 = load_frag(Bb, STR, waveN * 32,      kk);
      v16bf b1 = load_frag(Bb, STR, waveN * 32 + 16, kk);
      c00 = wmma_bf16(a0, b0, c00);
      c01 = wmma_bf16(a0, b1, c01);
      c10 = wmma_bf16(a1, b0, c10);
      c11 = wmma_bf16(a1, b1, c11);
    }
  }

  const int mW = mBlock + waveM * 32;
  const int nW = nBlock + waveN * 32;
  if constexpr (OUT_F32) {
    float* Y = (float*)Yv;
    #pragma unroll
    for (int r = 0; r < 8; ++r) {
      int row0 = mW + r + 8 * hlf;
      Y[(long)row0 * N + nW + l15]             = c00[r];
      Y[(long)row0 * N + nW + 16 + l15]        = c01[r];
      Y[(long)(row0 + 16) * N + nW + l15]      = c10[r];
      Y[(long)(row0 + 16) * N + nW + 16 + l15] = c11[r];
    }
  } else {
    __bf16* Y = (__bf16*)Yv;
    #pragma unroll
    for (int r = 0; r < 8; ++r) {
      int row0 = mW + r + 8 * hlf;
      Y[(long)row0 * N + nW + l15]             = to_bf16(c00[r]);
      Y[(long)row0 * N + nW + 16 + l15]        = to_bf16(c01[r]);
      Y[(long)(row0 + 16) * N + nW + l15]      = to_bf16(c10[r]);
      Y[(long)(row0 + 16) * N + nW + 16 + l15] = to_bf16(c11[r]);
    }
  }
}

// ---------- fused flash attention over projected tensors ----------
// qp,kp,vp: bf16 [NB][SEQ][PD]; att: bf16 [NB][SEQ][PD]
__global__ __launch_bounds__(256) void flash_kernel(
    const __bf16* __restrict__ qp, const __bf16* __restrict__ kp,
    const __bf16* __restrict__ vp, const unsigned char* __restrict__ mask,
    __bf16* __restrict__ att) {
  constexpr int KT = 32;           // keys per step (= WMMA K for PV)
  constexpr int PH = 256;          // P half for V staging
  constexpr int KSTR = PD + 8;     // K buffer row stride (520, rows 16B aligned)
  constexpr int VSTR = KT + 8;     // Vt / Ps row stride (40)
  __shared__ __align__(16) __bf16 Ks[KT * KSTR];    // 33.3 KB: full K tile 32x512
  __shared__ __align__(16) __bf16 Vt[PH * VSTR];    // 20.5 KB: transposed V half
  __shared__ __align__(16) __bf16 Ps[8][16][VSTR];  // 10.2 KB: per-wave probs

  const int tid = threadIdx.x;
  const int wave = tid >> 5, lane = tid & 31;
  const int hlf = lane >> 4, l15 = lane & 15;
  const int b = blockIdx.y;
  const int qRow = blockIdx.x * 128 + wave * 16;
  const float scale = 0.044194173824159216f;  // 1/sqrt(512)

  auto stageK = [&](int key0) {                // async: overlaps with PV WMMAs
    #pragma unroll
    for (int c = tid; c < (KT * PD) / 8; c += 256) {
      int key = c >> 6, p8 = (c & 63) * 8;
      async_cp16(&Ks[key * KSTR + p8],
                 kp + ((long)b * SEQ + key0 + key) * PD + p8);
    }
  };

  stageK(0);  // prologue: first key tile in flight while we load Q frags

  // resident Q fragments: 16 rows x P=512 -> 16 frags (128 VGPRs), reused 64x
  v16bf qf[16];
  #pragma unroll
  for (int i = 0; i < 16; ++i)
    qf[i] = load_frag(qp + ((long)b * SEQ + qRow) * PD, PD, 0, i * 32);

  float mrow[8], lsum[8];
  #pragma unroll
  for (int r = 0; r < 8; ++r) { mrow[r] = -__builtin_inff(); lsum[r] = 0.f; }
  v8f acc[32];
  #pragma unroll
  for (int t = 0; t < 32; ++t) acc[t] = (v8f){};

  for (int key0 = 0; key0 < SEQ; key0 += KT) {
    async_wait();
    __syncthreads();  // b1: K tile staged; prev step's Vt reads complete

    // ----- scores: s[16q x 32keys] = qp * kp^T over P=512 -----
    v8f s0 = {}, s1 = {};
    #pragma unroll
    for (int kk = 0; kk < 16; ++kk) {         // B-frag rows indexed by key (N)
      v16bf a   = qf[kk];
      v16bf bk0 = load_frag(Ks, KSTR, 0,  kk * 32);
      v16bf bk1 = load_frag(Ks, KSTR, 16, kk * 32);
      s0 = wmma_bf16(a, bk0, s0);
      s1 = wmma_bf16(a, bk1, s1);
    }

    // ----- mask + scale + online softmax (C-layout: col=l15, row=r+8*hlf) ---
    unsigned char m0 = mask[(long)b * SEQ + key0 + l15];
    unsigned char m1 = mask[(long)b * SEQ + key0 + 16 + l15];
    #pragma unroll
    for (int r = 0; r < 8; ++r) {
      float v0 = m0 ? s0[r] * scale : -__builtin_inff();
      float v1 = m1 ? s1[r] * scale : -__builtin_inff();
      float mx = fmaxf(v0, v1);
      #pragma unroll
      for (int d = 1; d < 16; d <<= 1) mx = fmaxf(mx, __shfl_xor(mx, d, 32));
      float mnew = fmaxf(mrow[r], mx);
      float fsc  = __expf(mrow[r] - mnew);
      mrow[r] = mnew;
      float p0 = __expf(v0 - mnew);
      float p1 = __expf(v1 - mnew);
      float ps = p0 + p1;
      #pragma unroll
      for (int d = 1; d < 16; d <<= 1) ps += __shfl_xor(ps, d, 32);
      lsum[r] = lsum[r] * fsc + ps;
      #pragma unroll
      for (int t = 0; t < 32; ++t) acc[t][r] *= fsc;   // rescale output rows
      Ps[wave][r + 8 * hlf][l15]      = to_bf16(p0);   // C-layout -> LDS
      Ps[wave][r + 8 * hlf][16 + l15] = to_bf16(p1);
    }
    // reload probs in A-layout (wave-private patch; DS ops in-order per wave)
    v16bf pa = load_frag(&Ps[wave][0][0], VSTR, 0, 0);

    __syncthreads();  // b2: all score reads of Ks complete -> safe to refill

    // ----- PV: acc += probs[16x32] * vp[32 x P], P in two halves -----
    #pragma unroll
    for (int hP = 0; hP < 2; ++hP) {
      #pragma unroll
      for (int c = tid; c < (KT * PH) / 8; c += 256) {  // stage V transposed
        int key = c >> 5, p8 = (c & 31) * 8;
        uint4 qv = *(const uint4*)(vp + ((long)b * SEQ + key0 + key) * PD + hP * PH + p8);
        const __bf16* e = (const __bf16*)&qv;
        #pragma unroll
        for (int j = 0; j < 8; ++j) Vt[(p8 + j) * VSTR + key] = e[j];
      }
      if (hP == 0 && key0 + KT < SEQ) stageK(key0 + KT);  // async next K tile
      __syncthreads();  // b3 / b5: Vt ready
      #pragma unroll
      for (int t = 0; t < 16; ++t) {          // B-frag rows indexed by p (N)
        v16bf bv = load_frag(Vt, VSTR, t * 16, 0);
        acc[hP * 16 + t] = wmma_bf16(pa, bv, acc[hP * 16 + t]);
      }
      if (hP == 0) __syncthreads();  // b4: PV half0 done before Vt refill
    }
  }

  // ----- finalize: divide by row sums, store bf16 -----
  #pragma unroll
  for (int t = 0; t < 32; ++t) {
    #pragma unroll
    for (int r = 0; r < 8; ++r) {
      float o = acc[t][r] / lsum[r];
      long row = (long)b * SEQ + qRow + r + 8 * hlf;
      att[row * PD + t * 16 + l15] = to_bf16(o);
    }
  }
}

// ---------- host launcher ----------
extern "C" void kernel_launch(void* const* d_in, const int* in_sizes, int n_in,
                              void* d_out, int out_size, void* d_ws, size_t ws_size,
                              hipStream_t stream) {
  const float* q    = (const float*)d_in[0];
  const float* k    = (const float*)d_in[1];
  const float* v    = (const float*)d_in[2];
  const unsigned char* mask = (const unsigned char*)d_in[3];
  const float* Wq   = (const float*)d_in[4];
  const float* Wk   = (const float*)d_in[5];
  const float* Wv   = (const float*)d_in[6];
  const float* Wout = (const float*)d_in[7];

  const size_t WSZ = (size_t)PD * DM;         // 524288 weight elements
  const size_t RP  = (size_t)NB * SEQ * PD;   // 16.7M projected elements
  __bf16* WqB   = (__bf16*)d_ws;
  __bf16* WkB   = WqB + WSZ;
  __bf16* WvB   = WkB + WSZ;
  __bf16* WoutB = WvB + WSZ;
  __bf16* qpB   = WoutB + WSZ;
  __bf16* kpB   = qpB + RP;
  __bf16* vpB   = kpB + RP;
  __bf16* attB  = vpB + RP;                   // total ~132 MiB of ws

  const int n4 = (int)(WSZ / 4);
  cvt_f32_bf16<<<(n4 + 255) / 256, 256, 0, stream>>>(Wq,   WqB,   n4);
  cvt_f32_bf16<<<(n4 + 255) / 256, 256, 0, stream>>>(Wk,   WkB,   n4);
  cvt_f32_bf16<<<(n4 + 255) / 256, 256, 0, stream>>>(Wv,   WvB,   n4);
  cvt_f32_bf16<<<(n4 + 255) / 256, 256, 0, stream>>>(Wout, WoutB, n4);

  const int R = NB * SEQ;                     // 32768
  dim3 gP(PD / 64, R / 128);                  // (8, 256)
  gemm_xwT_kernel<float, false><<<gP, 256, 0, stream>>>(q, WqB, qpB, R, DM, PD);
  gemm_xwT_kernel<float, false><<<gP, 256, 0, stream>>>(k, WkB, kpB, R, DM, PD);
  gemm_xwT_kernel<float, false><<<gP, 256, 0, stream>>>(v, WvB, vpB, R, DM, PD);

  dim3 gF(SEQ / 128, NB);                     // (16, 16)
  flash_kernel<<<gF, 256, 0, stream>>>(qpB, kpB, vpB, mask, attB);

  dim3 gO(DM / 64, R / 128);                  // (16, 256)
  gemm_xwT_kernel<__bf16, true><<<gO, 256, 0, stream>>>(attB, WoutB, d_out, R, PD, DM);
}
